// VarianceAdaptor_17849884082421
// MI455X (gfx1250) — matrix operand order repaired and checked by reading, not compiled, not run on hardware
//
#include <hip/hip_runtime.h>
#include <hip/hip_bf16.h>
#include <stdint.h>

// ---------------- problem constants ----------------
#define B_      64
#define T_      512
#define H_      256
#define F_      256
#define KW      3
#define HIST_   16
#define NBINS_  256
#define MAXLEN_ 2048
#define CIN_PAUSE   272
#define CINP_PAUSE  288     // padded to multiple of 32 for WMMA K-chunks
#define LN_EPS_ 1e-5f

typedef __bf16 bf16;
typedef __attribute__((ext_vector_type(16))) __bf16 v16bf;
typedef __attribute__((ext_vector_type(8)))  __bf16 v8bf;
typedef __attribute__((ext_vector_type(8)))  float  v8f;
typedef __attribute__((ext_vector_type(4)))  int    v4i;

// ---------------- workspace layout (bytes) ----------------
#define OFF_XB   ((size_t)0)                         // B*T*H bf16      = 16,777,216
#define OFF_PB   ((size_t)16777216)                  // B*T*288 bf16    = 18,874,368
#define OFF_Y1   ((size_t)35651584)                  // B*T*F bf16      = 16,777,216
#define OFF_W1B  ((size_t)52428800)                  // <= 512 KB
#define OFF_W2B  ((size_t)52953088)                  // <= 512 KB
#define OFF_CUM  ((size_t)53477376)                  // B*T int32
#define OFF_H    ((size_t)0)                         // reuse xb+pb region after predictors

// d_out layout (floats): output | log_d | pitch | energy | pause | mel_len
#define OUT_MAIN_N   ((size_t)B_*MAXLEN_*H_)         // 33,554,432
#define PRED_OFF     OUT_MAIN_N
#define MEL_OFF      (OUT_MAIN_N + (size_t)4*B_*T_)

// ---------------- small device helpers ----------------
__device__ __forceinline__ float flexf(float x, float a, float b, float c, float d) {
    float t = powf(1.0f + powf(b * fabsf(x), c), 1.0f / c);
    return a * b * x / t + d * x;
}

// async global->LDS copy of 16 bytes (guarded; falls back to plain copy)
__device__ __forceinline__ void g2l_copy16(bf16* dst, const bf16* src) {
#if __has_builtin(__builtin_amdgcn_global_load_async_to_lds_b128)
    __builtin_amdgcn_global_load_async_to_lds_b128(
        (__attribute__((address_space(1))) v4i*)(v4i*)src,
        (__attribute__((address_space(3))) v4i*)(v4i*)dst, 0, 0);
#else
    *(float4*)dst = *(const float4*)src;
#endif
}
__device__ __forceinline__ void g2l_wait() {
#if __has_builtin(__builtin_amdgcn_s_wait_asynccnt)
    __builtin_amdgcn_s_wait_asynccnt(0);
#endif
}

// ---------------- kernels ----------------

// convert x -> bf16 and build padded pause input (x | text_history | zeros)
__global__ __launch_bounds__(288)
void pack_inputs(const float* __restrict__ x, const float* __restrict__ textEmb,
                 bf16* __restrict__ xb, bf16* __restrict__ pbuf) {
    int g = blockIdx.x;                 // row = b*T + t
    int b = g / T_, t = g - b * T_;
    int f = threadIdx.x;
    if (f < H_) {
        bf16 v = (bf16)x[(size_t)g * H_ + f];
        xb[(size_t)g * H_ + f] = v;
        pbuf[(size_t)g * CINP_PAUSE + f] = v;
    } else if (f < H_ + HIST_) {
        int j = f - H_;
        int src = t - j;
        float val = (src >= 0) ? textEmb[b * T_ + src] : -1.0f;
        pbuf[(size_t)g * CINP_PAUSE + f] = (bf16)val;
    } else {
        pbuf[(size_t)g * CINP_PAUSE + f] = (bf16)0.0f;
    }
}

// Weights (KW, cin, F) fp32 -> bf16 swizzled into WMMA-B-fragment-ready layout:
//   d[((ch*2+hi) << 12) + (n << 4) + ii]  =  B[kk = ch*32 + hi*16 + ii][n]
// where kk indexes the (zero-padded) 3*cinp contraction axis. Each lane's whole
// 16-half B fragment is then one aligned 32-byte contiguous block.
__global__ void cvt_w_swz(const float* __restrict__ s, bf16* __restrict__ d,
                          int cin, int cinp) {
    int i = blockIdx.x * blockDim.x + threadIdx.x;
    if (i >= KW * cinp * F_) return;
    int ii = i & 15;
    int n  = (i >> 4) & (F_ - 1);
    int hb = (i >> 12) & 1;
    int ch = i >> 13;
    int kk = ch * 32 + hb * 16 + ii;
    int k  = kk / cinp;
    int c  = kk - k * cinp;
    d[i] = (c < cin) ? (bf16)s[((size_t)k * cin + c) * F_ + n] : (bf16)0.0f;
}

// Fused conv1d(K=3, SAME) as WMMA GEMM + bias + flex + LayerNorm.
// mode 0: write bf16 rows (next layer input).  mode 1: dot with wl, +bl, mask -> pred.
// One block = 16 time rows x 256 features. 256 threads = 8 waves, 2 col-tiles/wave.
template <int CINP>
__global__ __launch_bounds__(256)
void conv_block(const bf16* __restrict__ inp,
                const bf16* __restrict__ wsw,          // swizzled (see cvt_w_swz)
                const float* __restrict__ bias,
                const float* __restrict__ pfa, const float* __restrict__ pfb,
                const float* __restrict__ pfc, const float* __restrict__ pfd,
                const float* __restrict__ gam, const float* __restrict__ bet,
                int mode,
                bf16* __restrict__ out_bf,
                const float* __restrict__ wl, const float* __restrict__ bl,
                const unsigned char* __restrict__ mask,
                float* __restrict__ pred) {
    constexpr int CPK = CINP / 32;          // K-chunks per conv tap
    __shared__ __align__(16) bf16 sIn[18 * CINP];
    __shared__ float sy[16 * F_];
    __shared__ float sred[2][16][16];
    __shared__ float smv[2][16];

    const int tid  = threadIdx.x;
    const int lane = tid & 31;
    const int wv   = tid >> 5;
    const int g0   = blockIdx.x * 16;       // first global row of this tile
    const int b    = g0 / T_;
    const int t0   = g0 - b * T_;

    const float fa = *pfa, fbp = *pfb, fc = *pfc, fd = *pfd;

    // ---- stage input rows t0-1 .. t0+16 into LDS (zero-padded at batch edges)
    constexpr int VPR  = CINP / 8;          // 16B vectors per row
    constexpr int NVEC = 18 * VPR;
    for (int v = tid; v < NVEC; v += 256) {
        int row = v / VPR;
        int cv  = v - row * VPR;
        int t   = t0 + row - 1;
        bf16* dst = &sIn[row * CINP + cv * 8];
        if (t >= 0 && t < T_) {
            g2l_copy16(dst, &inp[((size_t)(b * T_ + t)) * CINP + cv * 8]);
        } else {
            *(float4*)dst = make_float4(0.f, 0.f, 0.f, 0.f);
        }
    }
    g2l_wait();
    __syncthreads();

    // ---- GEMM over K = 3*CINP in chunks of 32 (fully unrolled)
    v8f acc0 = {}; v8f acc1 = {};
    const int fb0 = wv * 32;
    const int fb1 = fb0 + 16;
    const int col = lane & 15;
    const int hi  = lane >> 4;

#pragma unroll
    for (int k = 0; k < KW; ++k) {
#pragma unroll
        for (int cb = 0; cb < CPK; ++cb) {
            const int ch    = k * CPK + cb;
            const int cbase = cb << 5;

            // A fragment (16-bit A layout): two contiguous 16B runs per lane
            const bf16* arow = &sIn[(col + k) * CINP + cbase + (hi << 4)];
            v8bf alo = *(const v8bf*)arow;
            v8bf ahi = *(const v8bf*)(arow + 8);
            v16bf af = __builtin_shufflevector(alo, ahi,
                        0, 1, 2, 3, 4, 5, 6, 7, 8, 9, 10, 11, 12, 13, 14, 15);

            // B fragments: one aligned 32B contiguous block per lane
            const bf16* pw = &wsw[(size_t)(ch * 2 + hi) << 12];
            __builtin_prefetch(pw + 8192, 0, 1);
            v16bf b0 = *(const v16bf*)&pw[(fb0 + col) << 4];
            v16bf b1 = *(const v16bf*)&pw[(fb1 + col) << 4];

            acc0 = __builtin_amdgcn_wmma_f32_16x16x32_bf16(false, af, false, b0,
                                                           (short)0, acc0, false, false);
            acc1 = __builtin_amdgcn_wmma_f32_16x16x32_bf16(false, af, false, b1,
                                                           (short)0, acc1, false, false);
        }
    }

    // ---- bias + flex into LDS tile (D layout: vgpr r -> m=r+8*hi, n=lane&15)
#pragma unroll
    for (int r = 0; r < 8; ++r) {
        int m  = r + (hi << 3);
        int f0 = fb0 + col;
        int f1 = fb1 + col;
        sy[m * F_ + f0] = flexf(acc0[r] + bias[f0], fa, fbp, fc, fd);
        sy[m * F_ + f1] = flexf(acc1[r] + bias[f1], fa, fbp, fc, fd);
    }
    __syncthreads();

    // ---- LayerNorm stats over F=256 per row (two-stage LDS reduction)
    {
        int row = tid >> 4, seg = tid & 15;
        float s = 0.f, s2 = 0.f;
#pragma unroll
        for (int j = 0; j < 16; ++j) {
            float v = sy[row * F_ + seg * 16 + j];
            s += v; s2 += v * v;
        }
        sred[0][row][seg] = s; sred[1][row][seg] = s2;
    }
    __syncthreads();
    if (tid < 16) {
        float s = 0.f, s2 = 0.f;
        for (int j = 0; j < 16; ++j) { s += sred[0][tid][j]; s2 += sred[1][tid][j]; }
        float mean = s * (1.0f / F_);
        float var  = s2 * (1.0f / F_) - mean * mean;
        smv[0][tid] = mean;
        smv[1][tid] = rsqrtf(var + LN_EPS_);
    }
    __syncthreads();

    const int row = tid >> 4, seg = tid & 15;
    const float mean = smv[0][row], inv = smv[1][row];
    if (mode == 0) {
#pragma unroll
        for (int j = 0; j < 16; ++j) {
            int f = seg * 16 + j;
            float y = (sy[row * F_ + f] - mean) * inv * gam[f] + bet[f];
            out_bf[(size_t)(g0 + row) * F_ + f] = (bf16)y;
        }
    } else {
        float part = 0.f;
#pragma unroll
        for (int j = 0; j < 16; ++j) {
            int f = seg * 16 + j;
            float y = (sy[row * F_ + f] - mean) * inv * gam[f] + bet[f];
            part += y * wl[f];
        }
        sred[0][row][seg] = part;
        __syncthreads();
        if (tid < 16) {
            float s = 0.f;
            for (int j = 0; j < 16; ++j) s += sred[0][tid][j];
            s += bl[0];
            int gr = g0 + tid;
            pred[gr] = mask[gr] ? 0.0f : s;
        }
    }
}

__device__ __forceinline__ int bucketize(float v, const float* __restrict__ bins) {
    // searchsorted(bins, v, side='left') over NBINS-1 = 255 bins
    int lo = 0, hi = NBINS_ - 1;
    while (lo < hi) {
        int mid = (lo + hi) >> 1;
        if (bins[mid] < v) lo = mid + 1; else hi = mid;
    }
    return lo;
}

__global__ __launch_bounds__(256)
void embed_sum(const float* __restrict__ x,
               const float* __restrict__ pt, const float* __restrict__ et,
               const float* __restrict__ at,
               const float* __restrict__ pbins, const float* __restrict__ ebins,
               const float* __restrict__ abins,
               const float* __restrict__ ptab, const float* __restrict__ etab,
               const float* __restrict__ atab,
               float* __restrict__ h) {
    int g = blockIdx.x;
    int f = threadIdx.x;
    int ip = bucketize(pt[g], pbins);
    int ie = bucketize(et[g], ebins);
    int ia = bucketize(at[g], abins);
    h[(size_t)g * H_ + f] = x[(size_t)g * H_ + f]
                          + ptab[ip * H_ + f] + etab[ie * H_ + f] + atab[ia * H_ + f];
}

__global__ void cumsum_dur(const int* __restrict__ dur, int* __restrict__ cum,
                           float* __restrict__ mel_out) {
    int b = blockIdx.x * blockDim.x + threadIdx.x;
    if (b >= B_) return;
    int run = 0;
    for (int t = 0; t < T_; ++t) { run += dur[b * T_ + t]; cum[b * T_ + t] = run; }
    mel_out[b] = (float)run;
}

__global__ __launch_bounds__(256)
void length_reg(const float* __restrict__ h, const int* __restrict__ cum,
                float* __restrict__ out) {
    int gb = blockIdx.x;                   // b*MAXLEN + t_out
    int b = gb / MAXLEN_, tout = gb - b * MAXLEN_;
    const int* c = &cum[b * T_];
    int mel = c[T_ - 1];
    int lo = 0, hi = T_;                   // searchsorted right: first c[i] > tout
    while (lo < hi) { int mid = (lo + hi) >> 1; if (c[mid] <= tout) lo = mid + 1; else hi = mid; }
    int idx = lo < (T_ - 1) ? lo : (T_ - 1);
    float v = (tout < mel) ? h[((size_t)(b * T_ + idx)) * H_ + threadIdx.x] : 0.0f;
    out[(size_t)gb * H_ + threadIdx.x] = v;
}

// ---------------- host launcher ----------------
extern "C" void kernel_launch(void* const* d_in, const int* in_sizes, int n_in,
                              void* d_out, int out_size, void* d_ws, size_t ws_size,
                              hipStream_t stream) {
    (void)in_sizes; (void)n_in; (void)out_size; (void)ws_size;

    // input leaf indices (setup_inputs insertion order, nested dicts recursed)
    const float*         x       = (const float*)d_in[0];
    const unsigned char* mask    = (const unsigned char*)d_in[1];
    const float*         textEmb = (const float*)d_in[2];
    const int*           dur     = (const int*)d_in[3];
    const float*         pt_t    = (const float*)d_in[4];
    const float*         et_t    = (const float*)d_in[5];
    const float*         pa_t    = (const float*)d_in[6];
    const float*         pbins   = (const float*)d_in[80];
    const float*         ebins   = (const float*)d_in[81];
    const float*         abins   = (const float*)d_in[82];
    const float*         ptab    = (const float*)d_in[83];
    const float*         etab    = (const float*)d_in[84];
    const float*         atab    = (const float*)d_in[85];

    char* ws = (char*)d_ws;
    bf16* xb   = (bf16*)(ws + OFF_XB);
    bf16* pbuf = (bf16*)(ws + OFF_PB);
    bf16* y1   = (bf16*)(ws + OFF_Y1);
    bf16* w1b  = (bf16*)(ws + OFF_W1B);
    bf16* w2b  = (bf16*)(ws + OFF_W2B);
    int*  cum  = (int*)(ws + OFF_CUM);
    float* hbuf = (float*)(ws + OFF_H);

    float* out = (float*)d_out;

    pack_inputs<<<B_ * T_, 288, 0, stream>>>(x, textEmb, xb, pbuf);

    const int nblk = (B_ * T_) / 16;  // 2048
    for (int p = 0; p < 4; ++p) {
        int base = 8 + 18 * p;
        const float* w1  = (const float*)d_in[base + 0];
        const float* b1  = (const float*)d_in[base + 1];
        const float* fa1 = (const float*)d_in[base + 2];
        const float* fbb1= (const float*)d_in[base + 3];
        const float* fc1 = (const float*)d_in[base + 4];
        const float* fd1 = (const float*)d_in[base + 5];
        const float* g1  = (const float*)d_in[base + 6];
        const float* be1 = (const float*)d_in[base + 7];
        const float* w2  = (const float*)d_in[base + 8];
        const float* b2  = (const float*)d_in[base + 9];
        const float* fa2 = (const float*)d_in[base + 10];
        const float* fbb2= (const float*)d_in[base + 11];
        const float* fc2 = (const float*)d_in[base + 12];
        const float* fd2 = (const float*)d_in[base + 13];
        const float* g2  = (const float*)d_in[base + 14];
        const float* be2 = (const float*)d_in[base + 15];
        const float* wl  = (const float*)d_in[base + 16];
        const float* bl  = (const float*)d_in[base + 17];

        // swizzled bf16 weights
        if (p < 3) {
            int n = KW * H_ * F_;
            cvt_w_swz<<<(n + 255) / 256, 256, 0, stream>>>(w1, w1b, H_, H_);
        } else {
            int n = KW * CINP_PAUSE * F_;
            cvt_w_swz<<<(n + 255) / 256, 256, 0, stream>>>(w1, w1b, CIN_PAUSE, CINP_PAUSE);
        }
        {
            int n = KW * F_ * F_;
            cvt_w_swz<<<(n + 255) / 256, 256, 0, stream>>>(w2, w2b, F_, F_);
        }

        if (p < 3) {
            conv_block<H_><<<nblk, 256, 0, stream>>>(xb, w1b, b1,
                                                     fa1, fbb1, fc1, fd1, g1, be1,
                                                     0, y1, nullptr, nullptr, nullptr, nullptr);
        } else {
            conv_block<CINP_PAUSE><<<nblk, 256, 0, stream>>>(pbuf, w1b, b1,
                                                     fa1, fbb1, fc1, fd1, g1, be1,
                                                     0, y1, nullptr, nullptr, nullptr, nullptr);
        }
        conv_block<F_><<<nblk, 256, 0, stream>>>(y1, w2b, b2,
                                                 fa2, fbb2, fc2, fd2, g2, be2,
                                                 1, nullptr, wl, bl, mask,
                                                 out + PRED_OFF + (size_t)p * (B_ * T_));
    }

    // h = x + pitch_emb + energy_emb + pause_emb   (reuses bf16 staging space)
    embed_sum<<<B_ * T_, 256, 0, stream>>>(x, pt_t, et_t, pa_t,
                                           pbins, ebins, abins,
                                           ptab, etab, atab, hbuf);
    cumsum_dur<<<1, 64, 0, stream>>>(dur, cum, out + MEL_OFF);
    length_reg<<<B_ * MAXLEN_, 256, 0, stream>>>(hbuf, cum, out);
}
